// MultiHeadAttention_67018669687544
// MI455X (gfx1250) — matrix operand order repaired and checked by reading, not compiled, not run on hardware
//
#include <hip/hip_runtime.h>
#include <hip/hip_bf16.h>

// MHA forward for MI455X (gfx1250, wave32, WMMA).
// bf16 WMMA everywhere; scores+softmax+PV fused through 256KB of the WGP's
// 320KB LDS so the 537MB attention-weights tensor is written to HBM exactly
// once (NT stores). All GEMM operands are L2-resident (192MB L2).

#define DIM   1024
#define NHEAD 16
#define HD    64
#define SEQ   2048
#define BATCH 2
#define BH    (BATCH * NHEAD)

// LDS score panel: 32 rows, row stride 2052 dwords (pad => stride % 64 == 4,
// conflict-free for phase-1 b32 stores, phase-2 b64 loads, phase-3 b128 loads).
#define LROW_F32 2052
#define LROW_BF  4104
#define LDS_BYTES (32 * LROW_F32 * 4)   // 262,656 B (< 320KB per workgroup)

typedef __attribute__((ext_vector_type(16))) __bf16 v16bf;
typedef __attribute__((ext_vector_type(8)))  __bf16 v8bf;
typedef __attribute__((ext_vector_type(2)))  __bf16 v2bf;
typedef __attribute__((ext_vector_type(8)))  float  v8f;
typedef __attribute__((ext_vector_type(4)))  float  v4f;
typedef __attribute__((ext_vector_type(2)))  float  v2f;

static __device__ __forceinline__ v8f wmma_bf16(v16bf a, v16bf b, v8f c) {
  return __builtin_amdgcn_wmma_f32_16x16x32_bf16(
      false, a, false, b, (short)0, c, false, false);
}

// ---- bf16 fragment loaders (pure b128 loads) -------------------------------
static __device__ __forceinline__ v16bf load_a_bf(const __bf16* __restrict__ base,
                                                  int lda, int m, int k0, int half) {
  const __bf16* p = base + (size_t)m * lda + k0 + half * 8;
  v8bf lo = *(const v8bf*)(p);
  v8bf hi = *(const v8bf*)(p + 16);
  return __builtin_shufflevector(lo, hi, 0, 1, 2, 3, 4, 5, 6, 7,
                                 8, 9, 10, 11, 12, 13, 14, 15);
}
static __device__ __forceinline__ v16bf load_b_bf(const __bf16* __restrict__ base,
                                                  int ldb, int n, int k0, int half) {
  const __bf16* p = base + (size_t)n * ldb + k0 + half * 8;
  v8bf lo = *(const v8bf*)(p);
  v8bf hi = *(const v8bf*)(p + 16);
  return __builtin_shufflevector(lo, hi, 0, 1, 2, 3, 4, 5, 6, 7,
                                 8, 9, 10, 11, 12, 13, 14, 15);
}

// ---- one-time f32 -> bf16 pack ---------------------------------------------
__global__ void pack_bf16(const float* __restrict__ in, __bf16* __restrict__ out,
                          int n) {
  const int i = (blockIdx.x * blockDim.x + threadIdx.x) * 8;
  if (i >= n) return;
  v4f x0 = *(const v4f*)(in + i);
  v4f x1 = *(const v4f*)(in + i + 4);
  v8bf o;
#pragma unroll
  for (int j = 0; j < 4; ++j) {
    o[j]     = (__bf16)x0[j];
    o[4 + j] = (__bf16)x1[j];
  }
  *(v8bf*)(out + i) = o;
}

// ---- projection GEMM: C = X @ W^T + bias; 32M x 64N per wave ---------------
// mode 0: Q -> bf16 [B,H,S,HD] * 1/sqrt(HD); mode 1: K -> bf16 [B,H,S,HD]
// mode 2: V -> bf16 [B,H,HD,S] (transposed); mode 3: O -> f32 d_out [4096,1024]
__global__ void proj_gemm(const __bf16* __restrict__ X, const __bf16* __restrict__ W,
                          const float* __restrict__ bias, void* __restrict__ outp,
                          int mode, float scale) {
  const int lane = threadIdx.x & 31, wave = threadIdx.x >> 5;
  const int wg = blockIdx.x * 4 + wave;           // 2048 wave-tiles
  const int mtile = wg & 127;                     // 128 M-tiles of 32
  const int ngrp  = wg >> 7;                      // 16 N-groups of 64
  const int m0 = mtile * 32, n0 = ngrp * 64;
  const int half = lane >> 4, ln = lane & 15;

  v8f accL[4] = {}, accH[4] = {};
  for (int k0 = 0; k0 < DIM; k0 += 32) {
    v16bf a0 = load_a_bf(X, DIM, m0 + ln,      k0, half);
    v16bf a1 = load_a_bf(X, DIM, m0 + 16 + ln, k0, half);
#pragma unroll
    for (int t = 0; t < 4; ++t) {
      v16bf b = load_b_bf(W, DIM, n0 + t * 16 + ln, k0, half);
      accL[t] = wmma_bf16(a0, b, accL[t]);
      accH[t] = wmma_bf16(a1, b, accH[t]);
    }
  }

#pragma unroll
  for (int t = 0; t < 4; ++t) {
    const int n = n0 + t * 16 + ln;
    const float bv = bias[n];
#pragma unroll
    for (int v = 0; v < 8; ++v) {
#pragma unroll
      for (int g = 0; g < 2; ++g) {
        const int m = m0 + g * 16 + v + half * 8;   // C/D: row = v + 8*half
        const float val = ((g ? accH[t][v] : accL[t][v]) + bv) * scale;
        if (mode == 3) {
          ((float*)outp)[(size_t)m * DIM + n] = val;
        } else {
          const int b_ = m >> 11, s_ = m & (SEQ - 1);
          const int h_ = n >> 6,  d_ = n & (HD - 1);
          size_t idx;
          if (mode == 2) idx = ((size_t)(b_ * NHEAD + h_) * HD + d_) * SEQ + s_;
          else           idx = ((size_t)(b_ * NHEAD + h_) * SEQ + s_) * HD + d_;
          ((__bf16*)outp)[idx] = (__bf16)val;
        }
      }
    }
  }
}

// ---- fused attention: scores -> softmax -> PV, score panel kept in LDS -----
// One workgroup (8 waves) per (bh, 32-row q-block). The 537MB attention matrix
// is written to HBM exactly once (normalized, NT).
__global__ void fused_attn(const __bf16* __restrict__ Qh, const __bf16* __restrict__ Kh,
                           const __bf16* __restrict__ Vt, const int* __restrict__ mask,
                           float* __restrict__ attn, __bf16* __restrict__ ctx) {
  extern __shared__ char smem_raw[];
  float*  sS = (float*)smem_raw;          // 32 x LROW_F32 f32 score panel
  __bf16* sP = (__bf16*)smem_raw;         // bf16 P alias, row r at r*LROW_BF

  const int lane = threadIdx.x & 31, wave = threadIdx.x >> 5;
  const int qb = blockIdx.x & 63;         // 64 q-blocks of 32
  const int bh = blockIdx.x >> 6;         // 32 (b,h)
  const int q0 = qb * 32;
  const int half = lane >> 4, ln = lane & 15;
  const int b_ = bh >> 4, h_ = bh & 15;

  const __bf16* Qb = Qh + (size_t)bh * SEQ * HD;
  const __bf16* Kb = Kh + (size_t)bh * SEQ * HD;
  const __bf16* Vb = Vt + (size_t)bh * HD * SEQ;
  const int* mrow = mask + b_ * SEQ;

  // ---- Phase 1: scores for cols [wave*256, wave*256+256) via WMMA ----------
  const int w0 = wave * 256;
  for (int cg = 0; cg < 4; ++cg) {
    const int c0 = w0 + cg * 64;
    v8f accL[4] = {}, accH[4] = {};
#pragma unroll
    for (int kk = 0; kk < HD; kk += 32) {
      v16bf a0 = load_a_bf(Qb, HD, q0 + ln,      kk, half);
      v16bf a1 = load_a_bf(Qb, HD, q0 + 16 + ln, kk, half);
#pragma unroll
      for (int t = 0; t < 4; ++t) {
        v16bf b = load_b_bf(Kb, HD, c0 + t * 16 + ln, kk, half);
        accL[t] = wmma_bf16(a0, b, accL[t]);
        accH[t] = wmma_bf16(a1, b, accH[t]);
      }
    }
#pragma unroll
    for (int t = 0; t < 4; ++t) {
      const int col = c0 + t * 16 + ln;
      const bool pad = (mrow[col] == 0);
#pragma unroll
      for (int v = 0; v < 8; ++v) {
#pragma unroll
        for (int g = 0; g < 2; ++g) {
          const int ql = g * 16 + v + half * 8;
          sS[ql * LROW_F32 + col] = pad ? -1e9f : (g ? accH[t][v] : accL[t][v]);
        }
      }
    }
  }
  __syncthreads();

  // ---- Phase 2: softmax; wave owns rows wave*4..wave*4+3 -------------------
  for (int r = 0; r < 4; ++r) {
    const int ql = wave * 4 + r;
    float vx[32], vy[32];
    float m = -3.4e38f;
#pragma unroll
    for (int i = 0; i < 32; ++i) {
      v2f pr = *(const v2f*)&sS[ql * LROW_F32 + lane * 2 + i * 64];  // ds b64
      vx[i] = pr[0]; vy[i] = pr[1];
      m = fmaxf(m, fmaxf(vx[i], vy[i]));
    }
#pragma unroll
    for (int off = 16; off >= 1; off >>= 1) m = fmaxf(m, __shfl_xor(m, off, 32));
    float s = 0.f;
#pragma unroll
    for (int i = 0; i < 32; ++i) {
      vx[i] = __expf(vx[i] - m);
      vy[i] = __expf(vy[i] - m);
      s += vx[i] + vy[i];
    }
#pragma unroll
    for (int off = 16; off >= 1; off >>= 1) s += __shfl_xor(s, off, 32);
    const float inv = 1.0f / s;

    float* grow = attn + ((size_t)bh * SEQ + q0 + ql) * SEQ;
#pragma unroll
    for (int i = 0; i < 32; ++i) {
      v2f o; o[0] = vx[i] * inv; o[1] = vy[i] * inv;
      __builtin_nontemporal_store(o, (v2f*)&grow[lane * 2 + i * 64]);  // HBM x1
      v2bf pb; pb[0] = (__bf16)o[0]; pb[1] = (__bf16)o[1];
      *(v2bf*)&sP[ql * LROW_BF + lane * 2 + i * 64] = pb;              // ds b32
    }
  }
  __syncthreads();

  // ---- Phase 3: PV; one 16x16 ctx tile per wave (A = bf16 P from LDS) ------
  const int g3 = wave >> 2, t3 = wave & 3;
  v8f acc = {};
  for (int k0 = 0; k0 < SEQ; k0 += 32) {
    const __bf16* ap = sP + (g3 * 16 + ln) * LROW_BF + k0 + half * 8;
    v8bf lo = *(const v8bf*)(ap);                                      // ds b128
    v8bf hi = *(const v8bf*)(ap + 16);
    v16bf a = __builtin_shufflevector(lo, hi, 0, 1, 2, 3, 4, 5, 6, 7,
                                      8, 9, 10, 11, 12, 13, 14, 15);
    v16bf b = load_b_bf(Vb, SEQ, t3 * 16 + ln, k0, half);
    acc = wmma_bf16(a, b, acc);
  }
  const int d = t3 * 16 + ln;
#pragma unroll
  for (int v = 0; v < 8; ++v) {
    const int q = q0 + g3 * 16 + v + half * 8;
    ctx[((size_t)(b_ * SEQ + q)) * DIM + h_ * HD + d] = (__bf16)acc[v];
  }
}

extern "C" void kernel_launch(void* const* d_in, const int* in_sizes, int n_in,
                              void* d_out, int out_size, void* d_ws, size_t ws_size,
                              hipStream_t stream) {
  const float* query = (const float*)d_in[0];
  const float* key   = (const float*)d_in[1];
  const float* value = (const float*)d_in[2];
  const int*   mask  = (const int*)  d_in[3];
  const float* Wq = (const float*)d_in[4];
  const float* bq = (const float*)d_in[5];
  const float* Wk = (const float*)d_in[6];
  const float* bk = (const float*)d_in[7];
  const float* Wv = (const float*)d_in[8];
  const float* bv = (const float*)d_in[9];
  const float* Wo = (const float*)d_in[10];
  const float* bo = (const float*)d_in[11];

  float* out  = (float*)d_out;                        // [2, 2048, 1024]
  float* attn = out + (size_t)BATCH * SEQ * DIM;      // [2, 16, 2048, 2048]

  const size_t XN = (size_t)BATCH * SEQ * DIM;        // 4,194,304
  const size_t WN = (size_t)DIM * DIM;                // 1,048,576
  const size_t HN = (size_t)BH * SEQ * HD;            // 4,194,304
  __bf16* p   = (__bf16*)d_ws;                        // ~67 MB bf16 workspace
  __bf16* Xq  = p; p += XN;
  __bf16* Xk  = p; p += XN;
  __bf16* Xv  = p; p += XN;
  __bf16* Wqb = p; p += WN;
  __bf16* Wkb = p; p += WN;
  __bf16* Wvb = p; p += WN;
  __bf16* Wob = p; p += WN;
  __bf16* Qh  = p; p += HN;
  __bf16* Kh  = p; p += HN;
  __bf16* Vt  = p; p += HN;
  __bf16* ctx = p; p += XN;

  const dim3 blk(128);                                // 4 waves / block
  const float invScale = 0.125f;                      // 1/sqrt(64)
  const int xblk = (int)(XN / (128 * 8));             // 4096
  const int wblk = (int)(WN / (128 * 8));             // 1024

  pack_bf16<<<xblk, blk, 0, stream>>>(query, Xq, (int)XN);
  pack_bf16<<<xblk, blk, 0, stream>>>(key,   Xk, (int)XN);
  pack_bf16<<<xblk, blk, 0, stream>>>(value, Xv, (int)XN);
  pack_bf16<<<wblk, blk, 0, stream>>>(Wq, Wqb, (int)WN);
  pack_bf16<<<wblk, blk, 0, stream>>>(Wk, Wkb, (int)WN);
  pack_bf16<<<wblk, blk, 0, stream>>>(Wv, Wvb, (int)WN);
  pack_bf16<<<wblk, blk, 0, stream>>>(Wo, Wob, (int)WN);

  proj_gemm<<<512, blk, 0, stream>>>(Xq, Wqb, bq, Qh, /*mode=*/0, invScale);
  proj_gemm<<<512, blk, 0, stream>>>(Xk, Wkb, bk, Kh, /*mode=*/1, 1.0f);
  proj_gemm<<<512, blk, 0, stream>>>(Xv, Wvb, bv, Vt, /*mode=*/2, 1.0f);

  fused_attn<<<BH * (SEQ / 32), dim3(256), LDS_BYTES, stream>>>(
      Qh, Kh, Vt, mask, attn, ctx);

  proj_gemm<<<512, blk, 0, stream>>>(ctx, Wob, bo, out, /*mode=*/3, 1.0f);
}
